// Model_5557687681586
// MI455X (gfx1250) — compile-verified, matrix operand order checked by reading, not run on hardware
//
#include <hip/hip_runtime.h>
#include <hip/hip_bf16.h>

// ---------------------------------------------------------------------------
// Types for WMMA
// ---------------------------------------------------------------------------
typedef __attribute__((ext_vector_type(16))) _Float16 v16h;
typedef __attribute__((ext_vector_type(8)))  _Float16 v8h;
typedef __attribute__((ext_vector_type(8)))  float    v8f;

#define N_NODES   50000
#define NODE_DIM  128
#define HIDDEN    256
#define BLOCK_OUT 128
#define NUM_CLS   40
#define NUM_CLS_P 48      // padded to 3 WMMA n-tiles
#define N_EDGES   800000

// ---------------------------------------------------------------------------
// WMMA helpers
// ---------------------------------------------------------------------------
__device__ __forceinline__ v8f wmma_f16(v16h a, v16h b, v8f c) {
    // 8 args: (neg_a, A, neg_b, B, c_mod, C, reuse_a, reuse_b)
    return __builtin_amdgcn_wmma_f32_16x16x32_f16(false, a, false, b,
                                                  (short)0, c, false, false);
}

// A-fragment (16xK row tile, f16) from LDS, row-major with `stride` halfs/row.
// 16-bit A layout: lanes 0-15 (M=lane) hold K=kt*32+[0..7] then +[16..23];
// lanes 16-31 (M=lane-16) hold K=kt*32+[8..15] then +[24..31].
__device__ __forceinline__ v16h load_a_lds(const _Float16* sA, int stride,
                                           int lane, int kt) {
    const int m  = lane & 15;
    const int hi = lane >> 4;
    const _Float16* p = sA + m * stride + kt * 32 + hi * 8;
    union { v16h v; v8h h[2]; } u;
    u.h[0] = *(const v8h*)p;          // ds_load_b128
    u.h[1] = *(const v8h*)(p + 16);   // ds_load_b128
    return u.v;
}

// B-fragment from a pre-packed global weight buffer: tile_idx = nt*KT + kt,
// each tile is 32 lanes x 16 halfs, per-lane contiguous (coalesced b128 loads).
__device__ __forceinline__ v16h load_b_packed(const _Float16* P, int tile_idx,
                                              int lane) {
    return *(const v16h*)(P + ((long long)tile_idx * 32 + lane) * 16);
}

// ---------------------------------------------------------------------------
// Weight packing: fp32 row-major W[K][Nsrc] -> f16 packed B tiles (N padded)
// ---------------------------------------------------------------------------
__global__ void pack_w(const float* __restrict__ W, _Float16* __restrict__ P,
                       int K, int N, int Nsrc) {
    const int KT = K >> 5;                    // K/32 k-tiles
    const int total = (N >> 4) * KT * 32 * 16;
    int idx = blockIdx.x * 256 + threadIdx.x;
    if (idx >= total) return;
    const int j    = idx & 15;
    const int l    = (idx >> 4) & 31;
    const int tile = idx >> 9;
    const int kt   = tile % KT;
    const int nt   = tile / KT;
    const int n    = nt * 16 + (l & 15);
    const int hi   = l >> 4;
    const int k    = kt * 32 + hi * 8 + (j & 7) + ((j >> 3) << 4);
    const float v  = (n < Nsrc) ? W[(long long)k * Nsrc + n] : 0.0f;
    P[idx] = (_Float16)v;
}

__global__ void pad_bias(const float* __restrict__ b, float* __restrict__ bp,
                         int n, int np) {
    int i = blockIdx.x * 64 + threadIdx.x;
    if (i < np) bp[i] = (i < n) ? b[i] : 0.0f;
}

// ---------------------------------------------------------------------------
// Stage 1: per-edge concat-gather -> f16 LDS -> WMMA vs W1 -> ReLU
//          -> atomic scatter-add into agg[dst], counts into cnt[dst]
// One 16-edge tile per 128-thread block; 4 waves each own 4 of 16 n-tiles.
// ---------------------------------------------------------------------------
__global__ __launch_bounds__(128) void edge_mlp1(
    const float* __restrict__ x, const long long* __restrict__ ei,
    const _Float16* __restrict__ W1p, const float* __restrict__ b1,
    float* __restrict__ agg, float* __restrict__ cnt) {
    __shared__ __align__(32) _Float16 sA[16 * 256];
    __shared__ int sD[16];

    const int tid = threadIdx.x;
    const long long e0 = (long long)blockIdx.x * 16;

    // gather + f32->f16 convert into LDS: 8 threads per row, 32 cols each
    {
        const int row = tid >> 3, seg = tid & 7;
        const long long e = e0 + row;
        const long long node = (seg < 4) ? ei[e] : ei[(long long)N_EDGES + e];
        const float* src = x + node * NODE_DIM + (seg & 3) * 32;
        _Float16* dstp = sA + row * 256 + seg * 32;
#pragma unroll
        for (int i = 0; i < 32; ++i) dstp[i] = (_Float16)src[i];
    }
    if (tid < 16) {
        const int d = (int)ei[(long long)N_EDGES + e0 + tid];
        sD[tid] = d;
        atomicAdd(cnt + d, 1.0f);
    }
    __syncthreads();

    const int lane = tid & 31, wave = tid >> 5;
    const int m = lane & 15, hi = lane >> 4;

#pragma unroll
    for (int nti = 0; nti < 4; ++nti) {
        const int nt = wave * 4 + nti;
        v8f acc = {};
#pragma unroll
        for (int kt = 0; kt < 8; ++kt)
            acc = wmma_f16(load_a_lds(sA, 256, lane, kt),
                           load_b_packed(W1p, nt * 8 + kt, lane), acc);
        const int n = nt * 16 + m;
        const float bias = b1[n];
#pragma unroll
        for (int i = 0; i < 8; ++i) {
            float v = fmaxf(acc[i] + bias, 0.0f);
            const long long d = sD[i + 8 * hi];
            atomicAdd(agg + d * HIDDEN + n, v);   // global_atomic_add_f32
        }
    }
}

// ---------------------------------------------------------------------------
// Stage 2: mean = agg/max(cnt,1) -> f16 LDS -> WMMA vs W2 -> ReLU -> h (f16)
// One 16-node tile per block; 4 waves x 2 n-tiles cover N=128.
// ---------------------------------------------------------------------------
__global__ __launch_bounds__(128) void node_mlp2(
    const float* __restrict__ agg, const float* __restrict__ cnt,
    const _Float16* __restrict__ W2p, const float* __restrict__ b2,
    _Float16* __restrict__ h16) {
    __shared__ __align__(32) _Float16 sA[16 * 256];

    const int tid = threadIdx.x;
    const long long n0 = (long long)blockIdx.x * 16;
    {
        const int row = tid >> 3, seg = tid & 7;
        const long long node = n0 + row;
        const float c = cnt[node];
        const float inv = 1.0f / fmaxf(c, 1.0f);
        const float* src = agg + node * HIDDEN + seg * 32;
        _Float16* dstp = sA + row * 256 + seg * 32;
#pragma unroll
        for (int i = 0; i < 32; ++i) dstp[i] = (_Float16)(src[i] * inv);
    }
    __syncthreads();

    const int lane = tid & 31, wave = tid >> 5;
    const int m = lane & 15, hi = lane >> 4;

#pragma unroll
    for (int nti = 0; nti < 2; ++nti) {
        const int nt = wave * 2 + nti;
        v8f acc = {};
#pragma unroll
        for (int kt = 0; kt < 8; ++kt)
            acc = wmma_f16(load_a_lds(sA, 256, lane, kt),
                           load_b_packed(W2p, nt * 8 + kt, lane), acc);
        const int n = nt * 16 + m;
        const float bias = b2[n];
#pragma unroll
        for (int i = 0; i < 8; ++i) {
            const float v = fmaxf(acc[i] + bias, 0.0f);
            const long long node = n0 + i + 8 * hi;
            h16[node * BLOCK_OUT + n] = (_Float16)v;
        }
    }
}

// ---------------------------------------------------------------------------
// Stage 3 (fused head): gather h[src]|h[dst] (f16) -> WMMA vs W3 -> ReLU -> z
// (f16 in LDS) -> WMMA vs W4 (N padded to 48) -> +b4 -> log_softmax -> out
// ---------------------------------------------------------------------------
__global__ __launch_bounds__(128) void edge_head(
    const _Float16* __restrict__ h16, const long long* __restrict__ ei,
    const _Float16* __restrict__ W3p, const float* __restrict__ b3,
    const _Float16* __restrict__ W4p, const float* __restrict__ b4p,
    float* __restrict__ out) {
    __shared__ __align__(32) _Float16 sA[16 * 256];
    __shared__ __align__(32) _Float16 sZ[16 * 128];
    __shared__ float sL[16 * NUM_CLS_P];

    const int tid = threadIdx.x;
    const long long e0 = (long long)blockIdx.x * 16;
    {
        const int row = tid >> 3, seg = tid & 7;
        const long long e = e0 + row;
        const long long node = (seg < 4) ? ei[e] : ei[(long long)N_EDGES + e];
        const _Float16* src = h16 + node * BLOCK_OUT + (seg & 3) * 32;
        _Float16* dstp = sA + row * 256 + seg * 32;
#pragma unroll
        for (int i = 0; i < 32; ++i) dstp[i] = src[i];
    }
    __syncthreads();

    const int lane = tid & 31, wave = tid >> 5;
    const int m = lane & 15, hi = lane >> 4;

    // z = relu(A @ W3 + b3), stored row-major f16 in LDS
#pragma unroll
    for (int nti = 0; nti < 2; ++nti) {
        const int nt = wave * 2 + nti;
        v8f acc = {};
#pragma unroll
        for (int kt = 0; kt < 8; ++kt)
            acc = wmma_f16(load_a_lds(sA, 256, lane, kt),
                           load_b_packed(W3p, nt * 8 + kt, lane), acc);
        const int n = nt * 16 + m;
        const float bias = b3[n];
#pragma unroll
        for (int i = 0; i < 8; ++i) {
            const float v = fmaxf(acc[i] + bias, 0.0f);
            sZ[(i + 8 * hi) * 128 + n] = (_Float16)v;
        }
    }
    __syncthreads();

    // logits = z @ W4 + b4  (K=128, N=48 padded); wave-uniform branch keeps
    // EXEC all-ones around the WMMAs.
    if (wave < 3) {
        const int nt = wave;
        v8f acc = {};
#pragma unroll
        for (int kt = 0; kt < 4; ++kt)
            acc = wmma_f16(load_a_lds(sZ, 128, lane, kt),
                           load_b_packed(W4p, nt * 4 + kt, lane), acc);
        const int n = nt * 16 + m;
        const float bias = b4p[n];
#pragma unroll
        for (int i = 0; i < 8; ++i)
            sL[(i + 8 * hi) * NUM_CLS_P + n] = acc[i] + bias;
    }
    __syncthreads();

    // row-wise log_softmax over the 40 real classes
    if (tid < 16) {
        const float* L = sL + tid * NUM_CLS_P;
        float mx = L[0];
#pragma unroll
        for (int c = 1; c < NUM_CLS; ++c) mx = fmaxf(mx, L[c]);
        float s = 0.0f;
#pragma unroll
        for (int c = 0; c < NUM_CLS; ++c) s += __expf(L[c] - mx);
        const float lse = mx + __logf(s);
        float* o = out + (e0 + tid) * NUM_CLS;
#pragma unroll
        for (int c = 0; c < NUM_CLS; ++c) o[c] = L[c] - lse;
    }
}

// ---------------------------------------------------------------------------
// Host launcher
// ---------------------------------------------------------------------------
extern "C" void kernel_launch(void* const* d_in, const int* in_sizes, int n_in,
                              void* d_out, int out_size, void* d_ws,
                              size_t ws_size, hipStream_t stream) {
    (void)in_sizes; (void)n_in; (void)out_size; (void)ws_size;

    const float*     x  = (const float*)d_in[0];
    const long long* ei = (const long long*)d_in[1];   // int64 [2, E]
    const float*     W1 = (const float*)d_in[2];
    const float*     b1 = (const float*)d_in[3];
    const float*     W2 = (const float*)d_in[4];
    const float*     b2 = (const float*)d_in[5];
    const float*     W3 = (const float*)d_in[6];
    const float*     b3 = (const float*)d_in[7];
    const float*     W4 = (const float*)d_in[8];
    const float*     b4 = (const float*)d_in[9];
    float* out = (float*)d_out;

    // workspace layout (256B aligned except h16, which is only 64B aligned)
    char* ws = (char*)d_ws;
    _Float16* W1p = (_Float16*)(ws + 0);            // 256x256 f16 -> 131072 B
    _Float16* W2p = (_Float16*)(ws + 131072);       // 256x128 f16 ->  65536 B
    _Float16* W3p = (_Float16*)(ws + 196608);       // 256x128 f16 ->  65536 B
    _Float16* W4p = (_Float16*)(ws + 262144);       // 128x48  f16 ->  12288 B
    float*    b4p = (float*)   (ws + 274432);       // 48 f32
    float*    agg = (float*)   (ws + 274688);       // 50000x256 f32 -> 51.2 MB
    float*    cnt = (float*)   (ws + 51474688);     // 50000 f32
    _Float16* h16 = (_Float16*)(ws + 51674688);     // 50000x128 f16 -> 12.8 MB

    // pack weights to f16 WMMA-B tile layout (re-done each call: deterministic)
    pack_w<<<(256 * 256 + 255) / 256, 256, 0, stream>>>(W1, W1p, 256, 256, 256);
    pack_w<<<(256 * 128 + 255) / 256, 256, 0, stream>>>(W2, W2p, 256, 128, 128);
    pack_w<<<(256 * 128 + 255) / 256, 256, 0, stream>>>(W3, W3p, 256, 128, 128);
    pack_w<<<(128 * 48  + 255) / 256, 256, 0, stream>>>(W4, W4p, 128, 48, 40);
    pad_bias<<<1, 64, 0, stream>>>(b4, b4p, NUM_CLS, NUM_CLS_P);

    // zero scatter accumulators (stream-ordered; graph-capture safe)
    hipMemsetAsync(agg, 0, (size_t)N_NODES * HIDDEN * sizeof(float), stream);
    hipMemsetAsync(cnt, 0, (size_t)N_NODES * sizeof(float), stream);

    edge_mlp1<<<N_EDGES / 16, 128, 0, stream>>>(x, ei, W1p, b1, agg, cnt);
    node_mlp2<<<N_NODES / 16, 128, 0, stream>>>(agg, cnt, W2p, b2, h16);
    edge_head<<<N_EDGES / 16, 128, 0, stream>>>(h16, ei, W3p, b3, W4p, b4p, out);
}
